// FastGeometricProduct_78297253806678
// MI455X (gfx1250) — matrix operand order repaired and checked by reading, not compiled, not run on hardware
//
#include <hip/hip_runtime.h>

// ---------------------------------------------------------------------------
// Geometric-product linear layer (Cl(4,1)) as an f16 WMMA GEMM.
//   out[b,o,k] = sum_{i,m} x[b,i,m^k] * sign(m,m^k) * w[o,i,m]
// GEMM view:  A[(b,k),(i,m)] = x[b,i,m^k]*S[k][m]   (generated on the fly)
//             B[(i,m),o]     = w[o,i,m]             (pre-converted to f16)
// M=32768, N=256, K=8192, fp16 inputs / fp32 accumulate.
// B tiles are streamed into LDS by the Tensor Data Mover (double buffered).
// ---------------------------------------------------------------------------

typedef __attribute__((ext_vector_type(16))) _Float16 v16h;
typedef __attribute__((ext_vector_type(8)))  float    v8f;
typedef __attribute__((ext_vector_type(4)))  float    f32x4;
typedef __attribute__((ext_vector_type(4)))  unsigned u32x4;
typedef __attribute__((ext_vector_type(2)))  _Float16 h2;
typedef __attribute__((ext_vector_type(4)))  _Float16 h4;

typedef __attribute__((ext_vector_type(4))) unsigned tdm_g0_t;  // D# group 0
typedef __attribute__((ext_vector_type(8))) int      tdm_g1_t;  // D# group 1
typedef __attribute__((ext_vector_type(4))) int      tdm_g2_t;  // D# groups 2/3
typedef __attribute__((ext_vector_type(8))) int      tdm_g4_t;  // 6-arg extra group

#define NB      32              // blades
#define IN_DIM  256
#define OUT_DIM 256
#define KTOT    (IN_DIM * NB)   // 8192
#define KT      64              // K tile (2 input channels i per step)
#define NSTEPS  (KTOT / KT)     // 128
#define AS_LD   80              // padded LDS stride in halves (64 + 16 pad)
#define NW      8               // waves per workgroup (one batch row b each)
#define NCOL    128             // output columns per workgroup

// sign of e_m * e_{m^k} in Cl(4,1): swap-count parity + metric (e4^2 = -1)
__device__ __forceinline__ float gp_sign(int m, int k) {
    const int bb = m ^ k;
    int swaps = 0;
#pragma unroll
    for (int i = 0; i < 5; ++i) {
        const int mask_gt = (~((1 << (i + 1)) - 1)) & 31;
        swaps += ((bb >> i) & 1) * __popc(m & mask_gt);
    }
    float s = (swaps & 1) ? -1.0f : 1.0f;
    if (m & bb & 16) s = -s;
    return s;
}

union Frag { u32x4 u[2]; v16h h; };

// ---------------------------------------------------------------------------
// Pre-pass: convert w (f32, [Out][K]) -> f16 in workspace, same layout.
// ---------------------------------------------------------------------------
__global__ __launch_bounds__(256)
void w_to_f16(const float* __restrict__ w, _Float16* __restrict__ wh) {
    const size_t u = (size_t)blockIdx.x * 256 + threadIdx.x;   // float4 units
    const f32x4 v = *(const f32x4*)(w + u * 4);
    h4 hv; hv[0] = (_Float16)v[0]; hv[1] = (_Float16)v[1];
           hv[2] = (_Float16)v[2]; hv[3] = (_Float16)v[3];
    *(h4*)(wh + u * 4) = hv;
}

// ---------------------------------------------------------------------------
// TDM: load one B tile (128 rows x 64 halves, row stride 8192 halves) into
// LDS at lds_addr with 32B padding after every 128B row (-> 160B LDS stride).
// 2D tensor: D# groups 2/3 zero-filled (tile_dim2 = 0 -> unused).
// ---------------------------------------------------------------------------
__device__ __forceinline__ void tdm_load_btile(const _Float16* gsrc,
                                               unsigned lds_addr) {
    tdm_g0_t g0;
    g0[0] = 1u;                                   // count=1, user descriptor
    g0[1] = lds_addr;                             // LDS byte address
    const unsigned long long ga = (unsigned long long)(uintptr_t)gsrc;
    g0[2] = (unsigned)(ga & 0xFFFFFFFFu);         // global_addr[31:0]
    g0[3] = (unsigned)((ga >> 32) & 0x01FFFFFFu)  // global_addr[56:32]
          | 0x80000000u;                          // type=2 ("image")
    tdm_g1_t g1;
    g1[0] = (int)((1u << 16)      // data_size = 1 -> 2-byte elements
                | (1u << 20)      // pad_enable
                | (4u << 22)      // pad_interval: 32 DWORDs (=128B) per row
                | (7u << 25));    // pad_amount: 8 DWORDs (=32B)
    g1[1] = (int)(64u << 16);     // tensor_dim0 = 64 elements (low 16 bits)
    g1[2] = (int)(128u << 16);    // tensor_dim0 hi=0 | tensor_dim1 = 128
    g1[3] = (int)(64u << 16);     // tensor_dim1 hi=0 | tile_dim0 = 64
    g1[4] = (int)(128u);          // tile_dim1 = 128 | tile_dim2 = 0
    g1[5] = (int)(8192u);         // tensor_dim0_stride = 8192 elements
    g1[6] = 0;                    // stride0 hi | tensor_dim1_stride lo (unused)
    g1[7] = 0;
    const tdm_g2_t z4 = {0, 0, 0, 0};
    const tdm_g4_t z8 = {0, 0, 0, 0, 0, 0, 0, 0};
    __builtin_amdgcn_tensor_load_to_lds(g0, g1, z4, z4, z8, 0);
}

// ---------------------------------------------------------------------------
// Main kernel: TDM-fed, double-buffered B tiles.
// ---------------------------------------------------------------------------
__global__ __launch_bounds__(256)
void geo_gemm_wmma_tdm(const float* __restrict__ x,
                       const _Float16* __restrict__ wh,
                       float* __restrict__ out) {
    __shared__ __align__(16) _Float16 Bs[2][NCOL * AS_LD];   // 2 x 20 KB
    __shared__ __align__(16) _Float16 As[NW * NB * AS_LD];   // 40 KB
    __shared__ __align__(8)  _Float16 Sh[NB * NB];           // 2 KB

    const int tid  = threadIdx.x;
    const int lane = tid & 31;
    const int wv   = tid >> 5;
    const int hi   = lane >> 4;
    const int l16  = lane & 15;
    const int b    = blockIdx.y * NW + wv;
    const int o0   = blockIdx.x * NCOL;

    // ---- sign table S[k][m] = sign(m, m^k) ---------------------------------
    for (int e = tid; e < NB * NB; e += 256) {
        const int k = e >> 5, m = e & 31;
        Sh[e] = (_Float16)gp_sign(m, k);
    }

    const _Float16* wtile = wh + (size_t)o0 * KTOT;   // [128 rows][K halves]
    const unsigned  bs0   = (unsigned)(uintptr_t)&Bs[0][0];
    const unsigned  bs1   = (unsigned)(uintptr_t)&Bs[1][0];

    // prologue: kick off DMA of step 0 into buffer 0
    if (wv == 0) tdm_load_btile(wtile, bs0);

    __syncthreads();   // Sh ready (TDM completion handled via TENSORcnt)

    v8f acc[2][8];
#pragma unroll
    for (int mt = 0; mt < 2; ++mt)
#pragma unroll
        for (int nt = 0; nt < 8; ++nt) { v8f z = {}; acc[mt][nt] = z; }

    const float* xrow = x + (size_t)b * (IN_DIM * NB);
    _Float16* Aw = &As[wv * (NB * AS_LD)];

    for (int step = 0; step < NSTEPS; ++step) {
        const int buf = step & 1;
        const int i0  = step * 2;

        if (wv == 0) __builtin_amdgcn_s_wait_tensorcnt(0);  // Bs[buf] arrived
        __syncthreads();   // all waves past last read of Bs[buf^1]; Bs[buf] ok

        // overlap: DMA next tile into the other buffer while we compute
        if (wv == 0 && step + 1 < NSTEPS)
            tdm_load_btile(wtile + (size_t)(step + 1) * KT, buf ? bs0 : bs1);

        // per-lane slice of x: lane holds blade 'lane' of channels i0, i0+1
        const float xv0 = xrow[i0 * NB + lane];
        const float xv1 = xrow[i0 * NB + NB + lane];

        // ---- build signed XOR-permuted A tile (wave-local, no barrier) -----
        // A[k][il*32+m] = x[b, i0+il, m^k] * S[k][m]
#pragma unroll
        for (int il = 0; il < 2; ++il) {
            const float xv = il ? xv1 : xv0;
#pragma unroll
            for (int pp = 0; pp < 16; ++pp) {
                const int k = pp * 2 + hi;
                const int m = l16 * 2;
                const float s0 = __shfl(xv, m ^ k, 32);        // ds_bpermute
                const float s1 = __shfl(xv, (m + 1) ^ k, 32);
                h2 hx; hx[0] = (_Float16)s0; hx[1] = (_Float16)s1;
                const h2 hs = *(const h2*)&Sh[k * NB + m];
                const h2 r = hx * hs;
                *(h2*)(&Aw[k * AS_LD + il * NB + m]) = r;
            }
        }

        // ---- WMMA main body: 2 k-substeps x 2 M-tiles x 8 N-tiles ----------
        const _Float16* Bw = &Bs[buf][0];
#pragma unroll
        for (int ks = 0; ks < 2; ++ks) {
            Frag a[2];
#pragma unroll
            for (int mt = 0; mt < 2; ++mt) {
                const _Float16* ap = &Aw[(mt * 16 + l16) * AS_LD + ks * 32 + hi * 16];
                a[mt].u[0] = *(const u32x4*)ap;
                a[mt].u[1] = *(const u32x4*)(ap + 8);
            }
            Frag bf[2];
            {
                const _Float16* bp = &Bw[l16 * AS_LD + ks * 32 + hi * 16];
                bf[0].u[0] = *(const u32x4*)bp;
                bf[0].u[1] = *(const u32x4*)(bp + 8);
            }
#pragma unroll
            for (int nt = 0; nt < 8; ++nt) {
                if (nt < 7) {   // prefetch next B fragment (hide LDS latency)
                    const _Float16* bp =
                        &Bw[((nt + 1) * 16 + l16) * AS_LD + ks * 32 + hi * 16];
                    bf[(nt + 1) & 1].u[0] = *(const u32x4*)bp;
                    bf[(nt + 1) & 1].u[1] = *(const u32x4*)(bp + 8);
                }
                acc[0][nt] = __builtin_amdgcn_wmma_f32_16x16x32_f16(
                    false, a[0].h, false, bf[nt & 1].h, (short)0, acc[0][nt],
                    false, false);
                acc[1][nt] = __builtin_amdgcn_wmma_f32_16x16x32_f16(
                    false, a[1].h, false, bf[nt & 1].h, (short)0, acc[1][nt],
                    false, false);
            }
        }
    }

    // ---- store: C/D layout row = vgpr j + hi*8, col = l16 -------------------
    float* ob = out + (size_t)b * (OUT_DIM * NB);
#pragma unroll
    for (int mt = 0; mt < 2; ++mt)
#pragma unroll
        for (int nt = 0; nt < 8; ++nt) {
            const int o  = o0 + nt * 16 + l16;
            const int k0 = mt * 16 + hi * 8;
            float* op = ob + (size_t)o * NB + k0;
            f32x4 lo, hi4;
            lo[0]  = acc[mt][nt][0]; lo[1]  = acc[mt][nt][1];
            lo[2]  = acc[mt][nt][2]; lo[3]  = acc[mt][nt][3];
            hi4[0] = acc[mt][nt][4]; hi4[1] = acc[mt][nt][5];
            hi4[2] = acc[mt][nt][6]; hi4[3] = acc[mt][nt][7];
            *(f32x4*)op       = lo;
            *(f32x4*)(op + 4) = hi4;
        }
}

// ---------------------------------------------------------------------------
// Fallback (no workspace): self-contained kernel, w converted in-loop.
// ---------------------------------------------------------------------------
__global__ __launch_bounds__(256)
void geo_gemm_wmma_fused(const float* __restrict__ x,
                         const float* __restrict__ w,
                         float* __restrict__ out) {
    __shared__ __align__(16) _Float16 Bs[NCOL * AS_LD];
    __shared__ __align__(16) _Float16 As[NW * NB * AS_LD];
    __shared__ __align__(8)  _Float16 Sh[NB * NB];

    const int tid  = threadIdx.x;
    const int lane = tid & 31;
    const int wv   = tid >> 5;
    const int hi   = lane >> 4;
    const int l16  = lane & 15;
    const int b    = blockIdx.y * NW + wv;
    const int o0   = blockIdx.x * NCOL;

    for (int e = tid; e < NB * NB; e += 256) {
        const int k = e >> 5, m = e & 31;
        Sh[e] = (_Float16)gp_sign(m, k);
    }
    __syncthreads();

    v8f acc[2][8];
#pragma unroll
    for (int mt = 0; mt < 2; ++mt)
#pragma unroll
        for (int nt = 0; nt < 8; ++nt) { v8f z = {}; acc[mt][nt] = z; }

    const float* xrow  = x + (size_t)b * (IN_DIM * NB);
    const float* wbase = w + (size_t)o0 * KTOT;
    _Float16* Aw = &As[wv * (NB * AS_LD)];

    for (int step = 0; step < NSTEPS; ++step) {
        const int i0 = step * 2;
        __syncthreads();

        const float xv0 = xrow[i0 * NB + lane];
        const float xv1 = xrow[i0 * NB + NB + lane];

#pragma unroll
        for (int j = 0; j < 8; ++j) {
            const int u  = tid + j * 256;
            const int o  = u >> 4;
            const int wi = u & 15;
            const f32x4 v = *(const f32x4*)(wbase + (size_t)o * KTOT + i0 * NB + wi * 4);
            h4 hv; hv[0] = (_Float16)v[0]; hv[1] = (_Float16)v[1];
                   hv[2] = (_Float16)v[2]; hv[3] = (_Float16)v[3];
            *(h4*)(&Bs[o * AS_LD + wi * 4]) = hv;
        }

#pragma unroll
        for (int il = 0; il < 2; ++il) {
            const float xv = il ? xv1 : xv0;
#pragma unroll
            for (int pp = 0; pp < 16; ++pp) {
                const int k = pp * 2 + hi;
                const int m = l16 * 2;
                const float s0 = __shfl(xv, m ^ k, 32);
                const float s1 = __shfl(xv, (m + 1) ^ k, 32);
                h2 hx; hx[0] = (_Float16)s0; hx[1] = (_Float16)s1;
                const h2 hs = *(const h2*)&Sh[k * NB + m];
                const h2 r = hx * hs;
                *(h2*)(&Aw[k * AS_LD + il * NB + m]) = r;
            }
        }

        __syncthreads();

#pragma unroll
        for (int ks = 0; ks < 2; ++ks) {
            Frag a[2];
#pragma unroll
            for (int mt = 0; mt < 2; ++mt) {
                const _Float16* ap = &Aw[(mt * 16 + l16) * AS_LD + ks * 32 + hi * 16];
                a[mt].u[0] = *(const u32x4*)ap;
                a[mt].u[1] = *(const u32x4*)(ap + 8);
            }
#pragma unroll
            for (int nt = 0; nt < 8; ++nt) {
                Frag bf;
                const _Float16* bp = &Bs[(nt * 16 + l16) * AS_LD + ks * 32 + hi * 16];
                bf.u[0] = *(const u32x4*)bp;
                bf.u[1] = *(const u32x4*)(bp + 8);
                acc[0][nt] = __builtin_amdgcn_wmma_f32_16x16x32_f16(
                    false, a[0].h, false, bf.h, (short)0, acc[0][nt], false, false);
                acc[1][nt] = __builtin_amdgcn_wmma_f32_16x16x32_f16(
                    false, a[1].h, false, bf.h, (short)0, acc[1][nt], false, false);
            }
        }
    }

    float* ob = out + (size_t)b * (OUT_DIM * NB);
#pragma unroll
    for (int mt = 0; mt < 2; ++mt)
#pragma unroll
        for (int nt = 0; nt < 8; ++nt) {
            const int o  = o0 + nt * 16 + l16;
            const int k0 = mt * 16 + hi * 8;
            float* op = ob + (size_t)o * NB + k0;
            f32x4 lo, hi4;
            lo[0]  = acc[mt][nt][0]; lo[1]  = acc[mt][nt][1];
            lo[2]  = acc[mt][nt][2]; lo[3]  = acc[mt][nt][3];
            hi4[0] = acc[mt][nt][4]; hi4[1] = acc[mt][nt][5];
            hi4[2] = acc[mt][nt][6]; hi4[3] = acc[mt][nt][7];
            *(f32x4*)op       = lo;
            *(f32x4*)(op + 4) = hi4;
        }
}

extern "C" void kernel_launch(void* const* d_in, const int* in_sizes, int n_in,
                              void* d_out, int out_size, void* d_ws, size_t ws_size,
                              hipStream_t stream) {
    (void)in_sizes; (void)n_in; (void)out_size;
    const float* x = (const float*)d_in[0];     // [1024, 256, 32] f32
    const float* w = (const float*)d_in[1];     // [256, 256, 32] f32
    float* out = (float*)d_out;                 // [1024, 256, 32] f32

    const size_t wh_bytes = (size_t)OUT_DIM * KTOT * 2;   // 4 MB of f16
    dim3 grid(2, 1024 / NW, 1);
    dim3 block(256, 1, 1);

    if (d_ws != nullptr && ws_size >= wh_bytes) {
        _Float16* wh = (_Float16*)d_ws;
        // pre-pass: w f32 -> f16 ([Out][K] layout preserved)
        const int units = OUT_DIM * KTOT / 4;             // float4 units
        w_to_f16<<<units / 256, 256, 0, stream>>>(w, wh);
        geo_gemm_wmma_tdm<<<grid, block, 0, stream>>>(x, wh, out);
    } else {
        geo_gemm_wmma_fused<<<grid, block, 0, stream>>>(x, w, out);
    }
}
